// EncoderLayer_18751827214890
// MI455X (gfx1250) — matrix operand order repaired and checked by reading, not compiled, not run on hardware
//
#include <hip/hip_runtime.h>

// ---------------------------------------------------------------------------
// Types for CDNA5 WMMA (wave32, 16x16x32 bf16 -> f32)
// ---------------------------------------------------------------------------
typedef __bf16 bf16_t;
typedef __attribute__((ext_vector_type(16))) __bf16 v16bf;
typedef __attribute__((ext_vector_type(8)))  __bf16 v8bf;
typedef __attribute__((ext_vector_type(8)))  float  v8f;

#define TT 32      // T
#define CC 64      // C
#define NNODES 512 // N
#define BATCH 4    // B
#define GRID_BN (BATCH * NNODES) // 2048 tiles
#define KNBR 8
#define NHID 256

// transposed-bf16 weight region offsets (elements) inside d_ws
#define OFF_WQT 0
#define OFF_WKT 4096
#define OFF_WVT 8192
#define OFF_WOT 12288
#define OFF_WTT 16384   // 3 taps * 4096
#define OFF_WQS 28672
#define OFF_WKS 32768
#define OFF_WOS 36864
#define OFF_W1T 40960   // [256][64]
#define OFF_W2T 57344   // [64][256]
#define W_ELEMS 73728

// ---------------------------------------------------------------------------
// WMMA helpers
// ---------------------------------------------------------------------------
__device__ __forceinline__ v8f wmma_bf(v16bf a, v16bf b, v8f c) {
  // emits v_wmma_f32_16x16x32_bf16
  return __builtin_amdgcn_wmma_f32_16x16x32_bf16(false, a, false, b,
                                                 (short)0, c, false, false);
}

// A fragment: row-major [rows][ldk] bf16. Lane l: row = row0 + (l&15);
// elements 0..7 -> K = k0 + 8*(l>>4) + e ; elements 8..15 -> K += 16.
// Chunks with K >= kreal are zeroed (K-padding for K=16 operands).
__device__ __forceinline__ v16bf a_frag(const bf16_t* base, int row0, int k0,
                                        int ldk, int kreal, int lane) {
  int half = lane >> 4;
  int r = row0 + (lane & 15);
  int ka = k0 + 8 * half;
  int kb = k0 + 16 + 8 * half;
  union { v16bf v; v8bf h[2]; } u;
  v8bf z = {};
  u.h[0] = (ka < kreal) ? *(const v8bf*)(base + r * ldk + ka) : z;
  u.h[1] = (kb < kreal) ? *(const v8bf*)(base + r * ldk + kb) : z;
  return u.v;
}

// A fragment with row bounds (for the shifted conv taps): rows outside
// [0,rows) read as zero.
__device__ __forceinline__ v16bf a_frag_sh(const bf16_t* base, int row0, int k0,
                                           int ldk, int kreal, int rows, int lane) {
  int r = row0 + (lane & 15);
  if (r < 0 || r >= rows) { v16bf z = {}; return z; }
  int half = lane >> 4;
  int ka = k0 + 8 * half;
  int kb = k0 + 16 + 8 * half;
  union { v16bf v; v8bf h[2]; } u;
  v8bf z = {};
  u.h[0] = (ka < kreal) ? *(const v8bf*)(base + r * ldk + ka) : z;
  u.h[1] = (kb < kreal) ? *(const v8bf*)(base + r * ldk + kb) : z;
  return u.v;
}

// B fragment from transposed layout [N][ldk] (K contiguous per row):
// lane l: n = n0 + (l&15); elements 0..7 -> K = k0 + 8*(l>>4) + e ;
// elements 8..15 -> K += 16.
__device__ __forceinline__ v16bf b_frag(const bf16_t* baseT, int n0, int k0,
                                        int ldk, int kreal, int lane) {
  int n = n0 + (lane & 15);
  int ka = k0 + 8 * (lane >> 4);
  int kb = k0 + 16 + 8 * (lane >> 4);
  union { v16bf v; v8bf h[2]; } u;
  v8bf z = {};
  u.h[0] = (ka < kreal) ? *(const v8bf*)(baseT + n * ldk + ka) : z;
  u.h[1] = (kb < kreal) ? *(const v8bf*)(baseT + n * ldk + kb) : z;
  return u.v;
}

// D fragment scatter: VGPR r -> row m0 + r + 8*(l>=16), col n0 + (l&15)
__device__ __forceinline__ void d_store_bf(bf16_t* dst, int m0, int n0, int ldn,
                                           v8f acc, int lane) {
  int n = n0 + (lane & 15);
  int mb = m0 + ((lane >> 4) << 3);
#pragma unroll
  for (int r = 0; r < 8; ++r) dst[(mb + r) * ldn + n] = (bf16_t)acc[r];
}

// Per-row LayerNorm over C=64 (32 rows). outb may be nullptr.
__device__ __forceinline__ void ln_rows(const float* y, const float* g,
                                        const float* bta, float* outf,
                                        bf16_t* outb, int tid) {
  if (tid < TT) {
    const float* row = y + tid * CC;
    float s = 0.f, s2 = 0.f;
    for (int c = 0; c < CC; ++c) { float v = row[c]; s += v; s2 += v * v; }
    float mu = s * (1.f / CC);
    float var = s2 * (1.f / CC) - mu * mu;
    float inv = rsqrtf(var + 1e-6f);
    for (int c = 0; c < CC; ++c) {
      float v = (row[c] - mu) * inv * g[c] + bta[c];
      outf[tid * CC + c] = v;
      if (outb) outb[tid * CC + c] = (bf16_t)v;
    }
  }
}

// ---------------------------------------------------------------------------
// Kernel 0: transpose+convert all weights to bf16 [N][K] layout in ws
// ---------------------------------------------------------------------------
__device__ __forceinline__ void tpose(const float* src, bf16_t* dst, int Kd,
                                      int Nd, int local) {
  int n = local / Kd, k = local % Kd;
  dst[local] = (bf16_t)src[k * Nd + n];
}

__global__ void prep_weights(const float* Wq_t, const float* Wk_t,
                             const float* Wv_t, const float* Wo_t,
                             const float* Wt, const float* Wq_s,
                             const float* Wk_s, const float* Wo_s,
                             const float* W1, const float* W2, bf16_t* wsb) {
  int idx = blockIdx.x * 256 + threadIdx.x;
  if (idx >= W_ELEMS) return;
  if (idx < OFF_WKT)       tpose(Wq_t, wsb + OFF_WQT, 64, 64, idx - OFF_WQT);
  else if (idx < OFF_WVT)  tpose(Wk_t, wsb + OFF_WKT, 64, 64, idx - OFF_WKT);
  else if (idx < OFF_WOT)  tpose(Wv_t, wsb + OFF_WVT, 64, 64, idx - OFF_WVT);
  else if (idx < OFF_WTT)  tpose(Wo_t, wsb + OFF_WOT, 64, 64, idx - OFF_WOT);
  else if (idx < OFF_WQS) {
    int rel = idx - OFF_WTT; int j = rel >> 12; int local = rel & 4095;
    tpose(Wt + j * 4096, wsb + OFF_WTT + j * 4096, 64, 64, local);
  }
  else if (idx < OFF_WKS)  tpose(Wq_s, wsb + OFF_WQS, 64, 64, idx - OFF_WQS);
  else if (idx < OFF_WOS)  tpose(Wk_s, wsb + OFF_WKS, 64, 64, idx - OFF_WKS);
  else if (idx < OFF_W1T)  tpose(Wo_s, wsb + OFF_WOS, 64, 64, idx - OFF_WOS);
  else if (idx < OFF_W2T)  tpose(W1,   wsb + OFF_W1T, 64, 256, idx - OFF_W1T);
  else                     tpose(W2,   wsb + OFF_W2T, 256, 64, idx - OFF_W2T);
}

// ---------------------------------------------------------------------------
// Kernel 1: temporal attention + LN1.  One workgroup per (b,n) tile.
// ---------------------------------------------------------------------------
__global__ void __launch_bounds__(256)
temporal_kernel(const float* __restrict__ xg, const float* __restrict__ ln1_g,
                const float* __restrict__ ln1_b, const bf16_t* __restrict__ wsb,
                float* __restrict__ x1f, bf16_t* __restrict__ x1b) {
  __shared__ __align__(32) bf16_t xb[TT * CC];
  __shared__ __align__(32) float  xf[TT * CC];
  __shared__ __align__(32) bf16_t qh[4 * TT * 16];   // [h][t][d]
  __shared__ __align__(32) bf16_t kh[4 * TT * 16];   // [h][s][d] == B^T
  __shared__ __align__(32) bf16_t vTh[4 * 16 * TT];  // [h][d][t] == B^T
  __shared__ __align__(32) float  sc[4 * TT * TT];
  __shared__ __align__(32) bf16_t ab[4 * TT * TT];
  __shared__ __align__(32) bf16_t ob[TT * CC];
  __shared__ __align__(32) float  y[TT * CC];

  const int tid = threadIdx.x, lane = tid & 31, wid = tid >> 5;
  const size_t bn = blockIdx.x;
  const float* xt = xg + bn * (TT * CC);

  for (int i = tid; i < TT * CC; i += 256) {
    float v = xt[i]; xf[i] = v; xb[i] = (bf16_t)v;
  }
  __syncthreads();

  // --- q,k,v projections: X[32x64] @ W[64x64], one 16x16 tile per wave/weight
  {
    const int m0 = (wid & 1) * 16, n0 = (wid >> 1) * 16;
    const bf16_t* Ws[3] = {wsb + OFF_WQT, wsb + OFF_WKT, wsb + OFF_WVT};
    const int h = n0 >> 4, d = lane & 15, mb = m0 + ((lane >> 4) << 3);
    for (int w = 0; w < 3; ++w) {
      v8f acc = {};
#pragma unroll
      for (int kk = 0; kk < 64; kk += 32)
        acc = wmma_bf(a_frag(xb, m0, kk, CC, CC, lane),
                      b_frag(Ws[w], n0, kk, CC, CC, lane), acc);
#pragma unroll
      for (int r = 0; r < 8; ++r) {
        if (w == 0)      qh[h * 512 + (mb + r) * 16 + d] = (bf16_t)acc[r];
        else if (w == 1) kh[h * 512 + (mb + r) * 16 + d] = (bf16_t)acc[r];
        else             vTh[h * 512 + d * TT + (mb + r)] = (bf16_t)acc[r];
      }
    }
  }
  __syncthreads();

  // --- scores = q @ k^T (K=16 zero-padded to 32), scale + causal mask
  for (int i = 0; i < 2; ++i) {
    int id = wid * 2 + i;
    int h = id >> 2, m0 = ((id >> 1) & 1) * 16, n0 = (id & 1) * 16;
    v8f acc = {};
    acc = wmma_bf(a_frag(qh + h * 512, m0, 0, 16, 16, lane),
                  b_frag(kh + h * 512, n0, 0, 16, 16, lane), acc);
    int s = n0 + (lane & 15), mb = m0 + ((lane >> 4) << 3);
#pragma unroll
    for (int r = 0; r < 8; ++r) {
      int t = mb + r;
      sc[h * 1024 + t * TT + s] = (s <= t) ? acc[r] * 0.25f : -1e9f;
    }
  }
  __syncthreads();

  // --- softmax rows -> bf16 attn
  if (tid < 128) {
    int h = tid >> 5, t = tid & 31;
    float* row = sc + h * 1024 + t * TT;
    float m = row[0];
    for (int s = 1; s < TT; ++s) m = fmaxf(m, row[s]);
    float sum = 0.f;
    for (int s = 0; s < TT; ++s) sum += __expf(row[s] - m);
    float inv = 1.f / sum;
    for (int s = 0; s < TT; ++s)
      ab[h * 1024 + t * TT + s] = (bf16_t)(__expf(row[s] - m) * inv);
  }
  __syncthreads();

  // --- o = attn @ v (K=32), one tile per wave
  {
    int h = wid >> 1, m0 = (wid & 1) * 16;
    v8f acc = {};
    acc = wmma_bf(a_frag(ab + h * 1024, m0, 0, TT, TT, lane),
                  b_frag(vTh + h * 512, 0, 0, TT, TT, lane), acc);
    d_store_bf(ob, m0, h * 16, CC, acc, lane);
  }
  __syncthreads();

  // --- x + o @ Wo_t
  {
    const int m0 = (wid & 1) * 16, n0 = (wid >> 1) * 16;
    v8f acc = {};
#pragma unroll
    for (int kk = 0; kk < 64; kk += 32)
      acc = wmma_bf(a_frag(ob, m0, kk, CC, CC, lane),
                    b_frag(wsb + OFF_WOT, n0, kk, CC, CC, lane), acc);
    int n = n0 + (lane & 15), mb = m0 + ((lane >> 4) << 3);
#pragma unroll
    for (int r = 0; r < 8; ++r)
      y[(mb + r) * CC + n] = acc[r] + xf[(mb + r) * CC + n];
  }
  __syncthreads();

  ln_rows(y, ln1_g, ln1_b, x1f + bn * (TT * CC), x1b + bn * (TT * CC), tid);
}

// ---------------------------------------------------------------------------
// Kernel 2: spatial q/k projections + temporal conv (3 shifted GEMMs)
// ---------------------------------------------------------------------------
__global__ void __launch_bounds__(256)
spatial_proj_kernel(const bf16_t* __restrict__ x1b,
                    const bf16_t* __restrict__ wsb, bf16_t* __restrict__ qsb,
                    bf16_t* __restrict__ ksb, bf16_t* __restrict__ vtb) {
  __shared__ __align__(32) bf16_t xb1[TT * CC];
  const int tid = threadIdx.x, lane = tid & 31, wid = tid >> 5;
  const size_t bn = blockIdx.x;
  const bf16_t* xt = x1b + bn * (TT * CC);
  for (int i = tid; i < TT * CC; i += 256) xb1[i] = xt[i];
  __syncthreads();

  const int m0 = (wid & 1) * 16, n0 = (wid >> 1) * 16;

  // qs and ks
  {
    v8f aq = {}, ak = {};
#pragma unroll
    for (int kk = 0; kk < 64; kk += 32) {
      v16bf a = a_frag(xb1, m0, kk, CC, CC, lane);
      aq = wmma_bf(a, b_frag(wsb + OFF_WQS, n0, kk, CC, CC, lane), aq);
      ak = wmma_bf(a, b_frag(wsb + OFF_WKS, n0, kk, CC, CC, lane), ak);
    }
    d_store_bf(qsb + bn * (TT * CC), m0, n0, CC, aq, lane);
    d_store_bf(ksb + bn * (TT * CC), m0, n0, CC, ak, lane);
  }

  // conv: vt[t] = sum_j x1[t+j-1] @ Wt[j]  (SAME padding)
  {
    v8f acc = {};
#pragma unroll
    for (int j = 0; j < 3; ++j)
#pragma unroll
      for (int kk = 0; kk < 64; kk += 32)
        acc = wmma_bf(a_frag_sh(xb1, m0 + j - 1, kk, CC, CC, TT, lane),
                      b_frag(wsb + OFF_WTT + j * 4096, n0, kk, CC, CC, lane),
                      acc);
    d_store_bf(vtb + bn * (TT * CC), m0, n0, CC, acc, lane);
  }
}

// ---------------------------------------------------------------------------
// Kernel 3: neighbor-gather spatial attention + Wo_s + LN2
// ---------------------------------------------------------------------------
__global__ void __launch_bounds__(256)
spatial_attn_kernel(const int* __restrict__ nbr_idx,
                    const bf16_t* __restrict__ qsb,
                    const bf16_t* __restrict__ ksb,
                    const bf16_t* __restrict__ vtb,
                    const float* __restrict__ x1f,
                    const bf16_t* __restrict__ wsb,
                    const float* __restrict__ ln2_g,
                    const float* __restrict__ ln2_b, float* __restrict__ x2f,
                    bf16_t* __restrict__ x2b) {
  __shared__ __align__(32) bf16_t qsl[TT * CC];
  __shared__ __align__(32) bf16_t knb[KNBR * TT * CC];
  __shared__ __align__(32) bf16_t vnb[KNBR * TT * CC];
  __shared__ __align__(32) float  sat[KNBR * 128];  // [k][t*4+h]
  __shared__ __align__(32) bf16_t ogb[TT * CC];
  __shared__ __align__(32) float  x1l[TT * CC];
  __shared__ __align__(32) float  y2[TT * CC];
  __shared__ int nbrs[KNBR];

  const int tid = threadIdx.x, lane = tid & 31, wid = tid >> 5;
  const size_t bn = blockIdx.x;
  const int b = (int)(bn / NNODES), n = (int)(bn % NNODES);

  if (tid < KNBR) nbrs[tid] = nbr_idx[n * KNBR + tid];
  for (int i = tid; i < TT * CC; i += 256) {
    qsl[i] = qsb[bn * (TT * CC) + i];
    x1l[i] = x1f[bn * (TT * CC) + i];
  }
  __syncthreads();
  for (int i = tid; i < KNBR * TT * CC; i += 256) {
    int k = i >> 11, off = i & 2047;
    size_t src = ((size_t)b * NNODES + (size_t)nbrs[k]) * (TT * CC) + off;
    knb[i] = ksb[src];
    vnb[i] = vtb[src];
  }
  __syncthreads();

  // s[k][t][h] = q_s[t,h,:] . k_n[k,t,h,:] / 4
  for (int task = tid; task < KNBR * 128; task += 256) {
    int k = task >> 7, rem = task & 127;
    int t = rem >> 2, h = rem & 3;
    const bf16_t* qp = qsl + t * CC + h * 16;
    const bf16_t* kp = knb + k * (TT * CC) + t * CC + h * 16;
    float dot = 0.f;
#pragma unroll
    for (int d = 0; d < 16; ++d) dot += (float)qp[d] * (float)kp[d];
    sat[k * 128 + rem] = dot * 0.25f;
  }
  __syncthreads();

  // softmax over the 8 neighbors, then og = sum_k a[k] * v_n[k]
  if (tid < 128) {
    int t = tid >> 2, h = tid & 3;
    float vals[KNBR], m = -1e30f;
#pragma unroll
    for (int k = 0; k < KNBR; ++k) { vals[k] = sat[k * 128 + tid]; m = fmaxf(m, vals[k]); }
    float sum = 0.f;
#pragma unroll
    for (int k = 0; k < KNBR; ++k) { vals[k] = __expf(vals[k] - m); sum += vals[k]; }
    float inv = 1.f / sum;
#pragma unroll
    for (int d = 0; d < 16; ++d) {
      float acc = 0.f;
#pragma unroll
      for (int k = 0; k < KNBR; ++k)
        acc += vals[k] * (float)vnb[k * (TT * CC) + t * CC + h * 16 + d];
      ogb[t * CC + h * 16 + d] = (bf16_t)(acc * inv);
    }
  }
  __syncthreads();

  // x1 + og @ Wo_s
  {
    const int m0 = (wid & 1) * 16, n0 = (wid >> 1) * 16;
    v8f acc = {};
#pragma unroll
    for (int kk = 0; kk < 64; kk += 32)
      acc = wmma_bf(a_frag(ogb, m0, kk, CC, CC, lane),
                    b_frag(wsb + OFF_WOS, n0, kk, CC, CC, lane), acc);
    int nc = n0 + (lane & 15), mb = m0 + ((lane >> 4) << 3);
#pragma unroll
    for (int r = 0; r < 8; ++r)
      y2[(mb + r) * CC + nc] = acc[r] + x1l[(mb + r) * CC + nc];
  }
  __syncthreads();

  ln_rows(y2, ln2_g, ln2_b, x2f + bn * (TT * CC), x2b + bn * (TT * CC), tid);
}

// ---------------------------------------------------------------------------
// Kernel 4: FFN + LN3 -> out
// ---------------------------------------------------------------------------
__global__ void __launch_bounds__(256)
ffn_kernel(const bf16_t* __restrict__ x2b, const float* __restrict__ x2f,
           const bf16_t* __restrict__ wsb, const float* __restrict__ b1,
           const float* __restrict__ b2, const float* __restrict__ ln3_g,
           const float* __restrict__ ln3_b, float* __restrict__ out) {
  __shared__ __align__(32) bf16_t x2l[TT * CC];
  __shared__ __align__(32) float  x2fl[TT * CC];
  __shared__ __align__(32) bf16_t hbuf[TT * NHID];
  __shared__ __align__(32) float  y3[TT * CC];

  const int tid = threadIdx.x, lane = tid & 31, wid = tid >> 5;
  const size_t bn = blockIdx.x;
  for (int i = tid; i < TT * CC; i += 256) {
    x2l[i] = x2b[bn * (TT * CC) + i];
    x2fl[i] = x2f[bn * (TT * CC) + i];
  }
  __syncthreads();

  // h = relu(x2 @ W1 + b1) : 32 tiles of 16x16, 4 per wave
  for (int i = 0; i < 4; ++i) {
    int id = wid + 8 * i;
    int m0 = (id & 1) * 16, n0 = (id >> 1) * 16;
    v8f acc = {};
#pragma unroll
    for (int kk = 0; kk < 64; kk += 32)
      acc = wmma_bf(a_frag(x2l, m0, kk, CC, CC, lane),
                    b_frag(wsb + OFF_W1T, n0, kk, CC, CC, lane), acc);
    int nc = n0 + (lane & 15), mb = m0 + ((lane >> 4) << 3);
    float bias = b1[nc];
#pragma unroll
    for (int r = 0; r < 8; ++r) {
      float v = acc[r] + bias;
      hbuf[(mb + r) * NHID + nc] = (bf16_t)(v > 0.f ? v : 0.f);
    }
  }
  __syncthreads();

  // x2 + h @ W2 + b2 : K = 256 -> 8 WMMA per tile
  {
    const int m0 = (wid & 1) * 16, n0 = (wid >> 1) * 16;
    v8f acc = {};
#pragma unroll
    for (int kk = 0; kk < NHID; kk += 32)
      acc = wmma_bf(a_frag(hbuf, m0, kk, NHID, NHID, lane),
                    b_frag(wsb + OFF_W2T, n0, kk, NHID, NHID, lane), acc);
    int nc = n0 + (lane & 15), mb = m0 + ((lane >> 4) << 3);
    float bias = b2[nc];
#pragma unroll
    for (int r = 0; r < 8; ++r)
      y3[(mb + r) * CC + nc] = acc[r] + x2fl[(mb + r) * CC + nc] + bias;
  }
  __syncthreads();

  ln_rows(y3, ln3_g, ln3_b, out + bn * (TT * CC), nullptr, tid);
}

// ---------------------------------------------------------------------------
// launch
// ---------------------------------------------------------------------------
extern "C" void kernel_launch(void* const* d_in, const int* in_sizes, int n_in,
                              void* d_out, int out_size, void* d_ws,
                              size_t ws_size, hipStream_t stream) {
  const float* x     = (const float*)d_in[0];
  const int*   nbr   = (const int*)d_in[2];
  const float* Wq_t  = (const float*)d_in[3];
  const float* Wk_t  = (const float*)d_in[4];
  const float* Wv_t  = (const float*)d_in[5];
  const float* Wo_t  = (const float*)d_in[6];
  const float* ln1_g = (const float*)d_in[7];
  const float* ln1_b = (const float*)d_in[8];
  const float* Wt    = (const float*)d_in[9];
  const float* Wq_s  = (const float*)d_in[10];
  const float* Wk_s  = (const float*)d_in[11];
  const float* Wo_s  = (const float*)d_in[12];
  const float* ln2_g = (const float*)d_in[13];
  const float* ln2_b = (const float*)d_in[14];
  const float* W1    = (const float*)d_in[15];
  const float* b1    = (const float*)d_in[16];
  const float* W2    = (const float*)d_in[17];
  const float* b2    = (const float*)d_in[18];
  const float* ln3_g = (const float*)d_in[19];
  const float* ln3_b = (const float*)d_in[20];
  float* out = (float*)d_out;

  char* wsc = (char*)d_ws;
  const size_t TILE_F = (size_t)GRID_BN * TT * CC * sizeof(float);   // 16 MB
  const size_t TILE_B = (size_t)GRID_BN * TT * CC * sizeof(bf16_t);  //  8 MB
  size_t off = 0;
  bf16_t* wsb = (bf16_t*)(wsc + off); off += (size_t)W_ELEMS * sizeof(bf16_t);
  off = (off + 255) & ~(size_t)255;
  float*  x1f = (float*)(wsc + off);  off += TILE_F;
  bf16_t* x1b = (bf16_t*)(wsc + off); off += TILE_B;
  bf16_t* qsb = (bf16_t*)(wsc + off); off += TILE_B;
  bf16_t* ksb = (bf16_t*)(wsc + off); off += TILE_B;
  bf16_t* vtb = (bf16_t*)(wsc + off); off += TILE_B;
  float*  x2f = (float*)(wsc + off);  off += TILE_F;
  bf16_t* x2b = (bf16_t*)(wsc + off); off += TILE_B;

  prep_weights<<<dim3((W_ELEMS + 255) / 256), dim3(256), 0, stream>>>(
      Wq_t, Wk_t, Wv_t, Wo_t, Wt, Wq_s, Wk_s, Wo_s, W1, W2, wsb);
  temporal_kernel<<<dim3(GRID_BN), dim3(256), 0, stream>>>(
      x, ln1_g, ln1_b, wsb, x1f, x1b);
  spatial_proj_kernel<<<dim3(GRID_BN), dim3(256), 0, stream>>>(
      x1b, wsb, qsb, ksb, vtb);
  spatial_attn_kernel<<<dim3(GRID_BN), dim3(256), 0, stream>>>(
      nbr, qsb, ksb, vtb, x1f, wsb, ln2_g, ln2_b, x2f, x2b);
  ffn_kernel<<<dim3(GRID_BN), dim3(256), 0, stream>>>(
      x2b, x2f, wsb, b1, b2, ln3_g, ln3_b, out);
}
